// BlocksCoreVAE_42880953483352
// MI455X (gfx1250) — matrix operand order repaired and checked by reading, not compiled, not run on hardware
//
#include <hip/hip_runtime.h>

// ---------------------------------------------------------------------------
// CDNA5 (gfx1250) implementation of BlocksCoreVAE forward.
// Big GEMMs (projection + block-linears) use v_wmma_f32_16x16x32_f16
// (f16 inputs staged in LDS, f32 accumulation). The pipeline is HBM-bound
// (~1.5 GB @ 23.3 TB/s); f16 WMMA keeps compute well under that bound.
// GEMM staging is double-buffered in LDS with b128 global loads + prefetch.
// ---------------------------------------------------------------------------

typedef __attribute__((ext_vector_type(16))) _Float16 v16h;
typedef __attribute__((ext_vector_type(8)))  _Float16 v8h;
typedef __attribute__((ext_vector_type(8)))  float    v8f;

#define DDIM   512
#define KNODES 64
#define BATCH  256
#define XSTR   40   // LDS row stride in halves (padded to dodge bank conflicts)

#define WMMA_F16(A, B, C) \
  __builtin_amdgcn_wmma_f32_16x16x32_f16(false, (A), false, (B), (short)0, (C), false, false)

// A-fragment (16x32 f16): lane r=lane&15 holds row M=r.
// half=lane>>4: VGPR j<4 -> K=8*half+2j..+1 ; VGPR j>=4 -> K=16+8*half+2(j-4)..+1
__device__ __forceinline__ v16h load_a_frag(const _Float16* tile, int r, int half) {
  const unsigned* u = (const unsigned*)(tile + r * XSTR);
  union { unsigned w[8]; v16h h; } f;
#pragma unroll
  for (int j = 0; j < 4; ++j) { f.w[j] = u[4 * half + j]; f.w[4 + j] = u[8 + 4 * half + j]; }
  return f.h;
}

// B-fragment (32x16 f16): N=lane&15; same K order as A: VGPR j<4 -> K=8*half+2j..+1 ; j>=4 -> K=16+8*half+2(j-4)..+1
// Staged K-transposed in LDS: row = N, contiguous in K -> two runs of 4 contiguous dwords per lane.
__device__ __forceinline__ v16h load_b_frag(const _Float16* tileT, int r, int half) {
  const unsigned* u = (const unsigned*)(tileT + r * XSTR);
  union { unsigned w[8]; v16h h; } f;
#pragma unroll
  for (int j = 0; j < 4; ++j) { f.w[j] = u[4 * half + j]; f.w[4 + j] = u[8 + 4 * half + j]; }
  return f.h;
}

// Pack 8 f32 -> 8 f16 and store as a single ds_store_b128 (dst 16B aligned).
__device__ __forceinline__ void store_x8(_Float16* dst, float4 a, float4 b) {
  v8h h;
  h[0] = (_Float16)a.x; h[1] = (_Float16)a.y; h[2] = (_Float16)a.z; h[3] = (_Float16)a.w;
  h[4] = (_Float16)b.x; h[5] = (_Float16)b.y; h[6] = (_Float16)b.z; h[7] = (_Float16)b.w;
  *(v8h*)dst = h;
}

// ---------------------------------------------------------------------------
// CQ-attention: one block per batch. Computes A = S1@Q and Bv = (S1@S2^T)@C.
// ---------------------------------------------------------------------------
template <int LQ>
__global__ __launch_bounds__(256) void attn_kernel(
    const float* __restrict__ Cp,     // node encodings [B,64,512]
    const float* __restrict__ Qp,     // query sequence [B,LQ,512]
    const float* __restrict__ Cmask,  // [B,64]
    const float* __restrict__ Qmask,  // [B,LQ]
    const float* __restrict__ w4C, const float* __restrict__ w4Q,
    const float* __restrict__ w4mlu, const float* __restrict__ biasp,
    float* __restrict__ Abuf, float* __restrict__ Bvbuf) {
  constexpr int NQT = (LQ + 15) / 16;
  constexpr int LQP = NQT * 16;
  __shared__ __attribute__((aligned(16))) _Float16 S0[64][LQP];   // raw S -> S2
  __shared__ __attribute__((aligned(16))) _Float16 S1s[64][LQP];  // S1
  __shared__ __attribute__((aligned(16))) _Float16 Tb[64][64];    // S1@S2^T
  __shared__ float cw[64];
  __shared__ float qw[LQP];

  const int b = blockIdx.x;
  const int t = threadIdx.x;
  const int lane = t & 31, wv = t >> 5;
  const float* Cb = Cp + (size_t)b * 64 * DDIM;
  const float* Qb = Qp + (size_t)b * LQ * DDIM;

  // cw[c] = C[c,:] . w4C   (wave per group of rows, shuffle reduction)
  for (int c = wv; c < 64; c += 8) {
    float s = 0.f;
    for (int d = lane; d < DDIM; d += 32) s += Cb[c * DDIM + d] * w4C[d];
#pragma unroll
    for (int o = 16; o > 0; o >>= 1) s += __shfl_xor(s, o, 32);
    if (lane == 0) cw[c] = s;
  }
  // qw[q] = Q[q,:] . w4Q
  for (int q = wv; q < LQ; q += 8) {
    float s = 0.f;
    for (int d = lane; d < DDIM; d += 32) s += Qb[q * DDIM + d] * w4Q[d];
#pragma unroll
    for (int o = 16; o > 0; o >>= 1) s += __shfl_xor(s, o, 32);
    if (lane == 0) qw[q] = s;
  }
  __syncthreads();

  // S[c][q] = sum_d C[c,d]*w4mlu[d]*Q[q,d] + cw[c] + qw[q] + bias
  const float bias = biasp[0];
  const int tx = t & 15, ty = t >> 4;
  float acc[4][NQT];
#pragma unroll
  for (int i = 0; i < 4; ++i)
#pragma unroll
    for (int j = 0; j < NQT; ++j) acc[i][j] = 0.f;
  for (int d = 0; d < DDIM; ++d) {
    float wmv = w4mlu[d];
    float cv[4];
#pragma unroll
    for (int i = 0; i < 4; ++i) cv[i] = Cb[(ty + 16 * i) * DDIM + d] * wmv;
#pragma unroll
    for (int j = 0; j < NQT; ++j) {
      int q = tx + 16 * j;
      float qv = (q < LQ) ? Qb[q * DDIM + d] : 0.f;
#pragma unroll
      for (int i = 0; i < 4; ++i) acc[i][j] += cv[i] * qv;
    }
  }
#pragma unroll
  for (int j = 0; j < NQT; ++j) {
    int q = tx + 16 * j;
    if (q < LQ) {
#pragma unroll
      for (int i = 0; i < 4; ++i) {
        int c = ty + 16 * i;
        S0[c][q] = (_Float16)(acc[i][j] + cw[c] + qw[q] + bias);
      }
    }
  }
  __syncthreads();

  // S1 = softmax over q (rows), masked by Qmask
  for (int c = wv; c < 64; c += 8) {
    float mx = -3.0e38f;
    for (int q = lane; q < LQ; q += 32) {
      float v = (Qmask[(size_t)b * LQ + q] > 0.f) ? (float)S0[c][q] : -1e30f;
      mx = fmaxf(mx, v);
    }
#pragma unroll
    for (int o = 16; o > 0; o >>= 1) mx = fmaxf(mx, __shfl_xor(mx, o, 32));
    float sum = 0.f;
    for (int q = lane; q < LQ; q += 32) {
      float v = (Qmask[(size_t)b * LQ + q] > 0.f) ? (float)S0[c][q] : -1e30f;
      float e = __expf(v - mx);
      S1s[c][q] = (_Float16)e;
      sum += e;
    }
#pragma unroll
    for (int o = 16; o > 0; o >>= 1) sum += __shfl_xor(sum, o, 32);
    float inv = 1.f / sum;
    for (int q = lane; q < LQ; q += 32) S1s[c][q] = (_Float16)((float)S1s[c][q] * inv);
  }
  __syncthreads();

  // S2 = softmax over c (columns), masked by Cmask; written in place over S0
  for (int q = t; q < LQ; q += 256) {
    float mx = -3.0e38f;
    for (int c = 0; c < 64; ++c) {
      float v = (Cmask[(size_t)b * 64 + c] > 0.f) ? (float)S0[c][q] : -1e30f;
      mx = fmaxf(mx, v);
    }
    float sum = 0.f;
    for (int c = 0; c < 64; ++c) {
      float v = (Cmask[(size_t)b * 64 + c] > 0.f) ? (float)S0[c][q] : -1e30f;
      sum += __expf(v - mx);
    }
    float inv = 1.f / sum;
    for (int c = 0; c < 64; ++c) {
      float v = (Cmask[(size_t)b * 64 + c] > 0.f) ? (float)S0[c][q] : -1e30f;
      S0[c][q] = (_Float16)(__expf(v - mx) * inv);
    }
  }
  __syncthreads();

  // T[c][k] = sum_q S1[c][q] * S2[k][q]
  for (int e = t; e < 64 * 64; e += 256) {
    int c = e >> 6, k = e & 63;
    float s = 0.f;
    for (int q = 0; q < LQ; ++q) s += (float)S1s[c][q] * (float)S0[k][q];
    Tb[c][k] = (_Float16)s;
  }
  __syncthreads();

  // A = S1 @ Q, Bv = T @ C
#pragma unroll
  for (int i = 0; i < 4; ++i) {
    int c = ty + 16 * i;
    for (int j = 0; j < 32; ++j) {
      int d = tx + 16 * j;
      float a = 0.f, bv = 0.f;
      for (int q = 0; q < LQ; ++q) a += (float)S1s[c][q] * Qb[q * DDIM + d];
      for (int k = 0; k < 64; ++k) bv += (float)Tb[c][k] * Cb[k * DDIM + d];
      size_t o = ((size_t)b * 64 + c) * DDIM + d;
      Abuf[o] = a;
      Bvbuf[o] = bv;
    }
  }
}

// ---------------------------------------------------------------------------
// Projection: H[b] = [C, A, C*A, C*Bv] @ P   (per batch: 64x2048 @ 2048x512)
// Block tile: M=64 x N=128, 8 waves in 2x4, each wave 32x32 (2x2 WMMA tiles).
// Double-buffered LDS staging, b128 global loads, prefetch 2 chunks ahead.
// ---------------------------------------------------------------------------
__global__ __launch_bounds__(256) void proj_kernel(
    const float* __restrict__ nodeC, const float* __restrict__ Abuf,
    const float* __restrict__ Bvbuf, const float* __restrict__ P,
    float* __restrict__ H) {
  __shared__ __attribute__((aligned(16))) _Float16 Xs[2][64 * XSTR];
  __shared__ __attribute__((aligned(16))) _Float16 WsT[2][128 * XSTR];
  const int b = blockIdx.y;
  const int n0 = blockIdx.x * 128;
  const int t = threadIdx.x, lane = t & 31, half = lane >> 4, r = lane & 15;
  const int wv = t >> 5, wm = wv >> 2, wn = wv & 3;
  v8f z = {0.f, 0.f, 0.f, 0.f, 0.f, 0.f, 0.f, 0.f};
  v8f acc00 = z, acc01 = z, acc10 = z, acc11 = z;
  const int xrow = t >> 2, xcolb = (t & 3) * 8;

  auto stage = [&](int kc, int buf) {
    // X tile (64 x 32): concat [C, A, C*A, C*Bv] built on the fly, f16-packed.
    {
      int sel = (kc + xcolb) >> 9;
      int km = (kc + xcolb) & 511;
      size_t base = ((size_t)b * 64 + xrow) * DDIM + km;
      const float* p0 = (sel == 1) ? Abuf : nodeC;
      const float* p1 = (sel == 2) ? Abuf : Bvbuf;
      float4 a0 = *(const float4*)(p0 + base);
      float4 a1 = *(const float4*)(p0 + base + 4);
      if (sel >= 2) {
        float4 m0 = *(const float4*)(p1 + base);
        float4 m1 = *(const float4*)(p1 + base + 4);
        a0.x *= m0.x; a0.y *= m0.y; a0.z *= m0.z; a0.w *= m0.w;
        a1.x *= m1.x; a1.y *= m1.y; a1.z *= m1.z; a1.w *= m1.w;
      }
      store_x8(&Xs[buf][xrow * XSTR + xcolb], a0, a1);
    }
    // W tile (32 x 128) -> K-transposed LDS; b128 coalesced global loads.
#pragma unroll
    for (int ii = 0; ii < 4; ++ii) {
      int lin4 = ii * 256 + t;
      int kk = lin4 >> 5, n = (lin4 & 31) * 4;
      float4 w = *(const float4*)&P[(size_t)(kc + kk) * DDIM + n0 + n];
      WsT[buf][(n + 0) * XSTR + kk] = (_Float16)w.x;
      WsT[buf][(n + 1) * XSTR + kk] = (_Float16)w.y;
      WsT[buf][(n + 2) * XSTR + kk] = (_Float16)w.z;
      WsT[buf][(n + 3) * XSTR + kk] = (_Float16)w.w;
    }
    // Prefetch the chunk after this one (2 ahead of compute).
    if (kc + 32 < 2048)
      __builtin_prefetch(&P[(size_t)(kc + 32 + (t >> 3)) * DDIM + n0 + (t & 7) * 16], 0, 1);
  };

  stage(0, 0);
  __syncthreads();
  int cur = 0;
  for (int kc = 0; kc < 2048; kc += 32) {
    if (kc + 32 < 2048) stage(kc + 32, cur ^ 1);
    v16h a0 = load_a_frag(&Xs[cur][(32 * wm) * XSTR], r, half);
    v16h a1 = load_a_frag(&Xs[cur][(32 * wm + 16) * XSTR], r, half);
    v16h b0 = load_b_frag(&WsT[cur][(32 * wn) * XSTR], r, half);
    v16h b1 = load_b_frag(&WsT[cur][(32 * wn + 16) * XSTR], r, half);
    acc00 = WMMA_F16(a0, b0, acc00);
    acc01 = WMMA_F16(a0, b1, acc01);
    acc10 = WMMA_F16(a1, b0, acc10);
    acc11 = WMMA_F16(a1, b1, acc11);
    __syncthreads();
    cur ^= 1;
  }
#pragma unroll
  for (int i = 0; i < 8; ++i) {  // D layout: M = i + 8*half, N = r
    int mb = 32 * wm + i + 8 * half;
    int nb = n0 + 32 * wn + r;
    H[((size_t)b * 64 + mb) * DDIM + nb] = acc00[i];
    H[((size_t)b * 64 + mb) * DDIM + nb + 16] = acc01[i];
    H[((size_t)b * 64 + (mb + 16)) * DDIM + nb] = acc10[i];
    H[((size_t)b * 64 + (mb + 16)) * DDIM + nb + 16] = acc11[i];
  }
}

// ---------------------------------------------------------------------------
// Block linear: for node kn: out[:,kn,:] = concat(s0,s1,s2)[:,kn,:] @ W[kn] + b[kn]
// GEMM 256 x KD x 512 per kn. Grid: (N/128, M/64, 64). Double-buffered LDS.
// ---------------------------------------------------------------------------
__global__ __launch_bounds__(256) void blocklin_kernel(
    const float* __restrict__ s0, const float* __restrict__ s1,
    const float* __restrict__ s2, const float* __restrict__ W,
    const float* __restrict__ bias, float* __restrict__ out, int KD) {
  __shared__ __attribute__((aligned(16))) _Float16 Xs[2][64 * XSTR];
  __shared__ __attribute__((aligned(16))) _Float16 WsT[2][128 * XSTR];
  const int n0 = blockIdx.x * 128;
  const int m0 = blockIdx.y * 64;
  const int kn = blockIdx.z;
  const int t = threadIdx.x, lane = t & 31, half = lane >> 4, r = lane & 15;
  const int wv = t >> 5, wm = wv >> 2, wn = wv & 3;
  v8f z = {0.f, 0.f, 0.f, 0.f, 0.f, 0.f, 0.f, 0.f};
  v8f acc00 = z, acc01 = z, acc10 = z, acc11 = z;
  const int xrow = t >> 2, xcolb = (t & 3) * 8;

  auto stage = [&](int kc, int buf) {
    {  // X tile: rows = batches m0..m0+63; K = concat of up to 3 sources.
      int sel = (kc + xcolb) >> 9;
      int km = (kc + xcolb) & 511;
      const float* sp = (sel == 0) ? s0 : ((sel == 1) ? s1 : s2);
      size_t base = ((size_t)(m0 + xrow) * 64 + kn) * DDIM + km;
      float4 x0 = *(const float4*)(sp + base);
      float4 x1 = *(const float4*)(sp + base + 4);
      store_x8(&Xs[buf][xrow * XSTR + xcolb], x0, x1);
    }
#pragma unroll
    for (int ii = 0; ii < 4; ++ii) {
      int lin4 = ii * 256 + t;
      int kk = lin4 >> 5, n = (lin4 & 31) * 4;
      float4 w = *(const float4*)&W[((size_t)kn * KD + kc + kk) * DDIM + n0 + n];
      WsT[buf][(n + 0) * XSTR + kk] = (_Float16)w.x;
      WsT[buf][(n + 1) * XSTR + kk] = (_Float16)w.y;
      WsT[buf][(n + 2) * XSTR + kk] = (_Float16)w.z;
      WsT[buf][(n + 3) * XSTR + kk] = (_Float16)w.w;
    }
    if (kc + 32 < KD)
      __builtin_prefetch(&W[((size_t)kn * KD + kc + 32 + (t >> 3)) * DDIM + n0 + (t & 7) * 16], 0, 1);
  };

  stage(0, 0);
  __syncthreads();
  int cur = 0;
  for (int kc = 0; kc < KD; kc += 32) {
    if (kc + 32 < KD) stage(kc + 32, cur ^ 1);
    v16h a0 = load_a_frag(&Xs[cur][(32 * wm) * XSTR], r, half);
    v16h a1 = load_a_frag(&Xs[cur][(32 * wm + 16) * XSTR], r, half);
    v16h b0 = load_b_frag(&WsT[cur][(32 * wn) * XSTR], r, half);
    v16h b1 = load_b_frag(&WsT[cur][(32 * wn + 16) * XSTR], r, half);
    acc00 = WMMA_F16(a0, b0, acc00);
    acc01 = WMMA_F16(a0, b1, acc01);
    acc10 = WMMA_F16(a1, b0, acc10);
    acc11 = WMMA_F16(a1, b1, acc11);
    __syncthreads();
    cur ^= 1;
  }
#pragma unroll
  for (int i = 0; i < 8; ++i) {
    int mb = m0 + 32 * wm + i + 8 * half;
    int nb = n0 + 32 * wn + r;
    out[((size_t)mb * 64 + kn) * DDIM + nb] = acc00[i] + bias[kn * DDIM + nb];
    out[((size_t)mb * 64 + kn) * DDIM + nb + 16] = acc01[i] + bias[kn * DDIM + nb + 16];
    out[((size_t)(mb + 16) * 64 + kn) * DDIM + nb] = acc10[i] + bias[kn * DDIM + nb];
    out[((size_t)(mb + 16) * 64 + kn) * DDIM + nb + 16] = acc11[i] + bias[kn * DDIM + nb + 16];
  }
}

// pred = eps * exp(0.5*logvar) + mu
__global__ __launch_bounds__(256) void vae_sample_kernel(
    const float* __restrict__ eps, const float* __restrict__ mu,
    const float* __restrict__ logvar, float* __restrict__ pred, int n) {
  int i = blockIdx.x * 256 + threadIdx.x;
  if (i < n) pred[i] = eps[i] * expf(0.5f * logvar[i]) + mu[i];
}

extern "C" void kernel_launch(void* const* d_in, const int* in_sizes, int n_in,
                              void* d_out, int out_size, void* d_ws, size_t ws_size,
                              hipStream_t stream) {
  (void)in_sizes; (void)n_in; (void)out_size; (void)ws_size;
  const float* act     = (const float*)d_in[0];
  const float* obs     = (const float*)d_in[1];
  const float* amask   = (const float*)d_in[2];
  const float* omask   = (const float*)d_in[3];
  const float* nodeC   = (const float*)d_in[4];
  const float* nmask   = (const float*)d_in[5];
  // d_in[6] node_embeddings: unused by the reference
  const float* w4C_o   = (const float*)d_in[7];
  const float* w4Q_o   = (const float*)d_in[8];
  const float* w4mlu_o = (const float*)d_in[9];
  const float* bias_o  = (const float*)d_in[10];
  const float* w4C_a   = (const float*)d_in[11];
  const float* w4Q_a   = (const float*)d_in[12];
  const float* w4mlu_a = (const float*)d_in[13];
  const float* bias_a  = (const float*)d_in[14];
  const float* obs_prj = (const float*)d_in[15];
  const float* act_prj = (const float*)d_in[16];
  const float* Wmu_pr  = (const float*)d_in[17];
  const float* bmu_pr  = (const float*)d_in[18];
  const float* Wsig_pr = (const float*)d_in[19];
  const float* bsig_pr = (const float*)d_in[20];
  const float* Wmu_po  = (const float*)d_in[21];
  const float* bmu_po  = (const float*)d_in[22];
  const float* Wsig_po = (const float*)d_in[23];
  const float* bsig_po = (const float*)d_in[24];
  const float* eps_pr  = (const float*)d_in[25];
  const float* eps_po  = (const float*)d_in[26];

  float* out = (float*)d_out;
  float* ws  = (float*)d_ws;
  const size_t TS = (size_t)BATCH * KNODES * DDIM;  // 8388608 elements
  float* A_a  = ws;            float* Bv_a = ws + TS;
  float* A_o  = ws + 2 * TS;   float* Bv_o = ws + 3 * TS;
  float* h_na = ws + 4 * TS;   float* h_no = ws + 5 * TS;

  // Attention (action: LQ=50, obs: LQ=200)
  attn_kernel<50><<<BATCH, 256, 0, stream>>>(nodeC, act, nmask, amask,
                                             w4C_a, w4Q_a, w4mlu_a, bias_a, A_a, Bv_a);
  attn_kernel<200><<<BATCH, 256, 0, stream>>>(nodeC, obs, nmask, omask,
                                              w4C_o, w4Q_o, w4mlu_o, bias_o, A_o, Bv_o);
  // Projections (WMMA)
  proj_kernel<<<dim3(4, BATCH), 256, 0, stream>>>(nodeC, A_a, Bv_a, act_prj, h_na);
  proj_kernel<<<dim3(4, BATCH), 256, 0, stream>>>(nodeC, A_o, Bv_o, obs_prj, h_no);
  // Block linears (WMMA): mu/logvar written straight into d_out slots
  dim3 g(4, 4, KNODES);
  blocklin_kernel<<<g, 256, 0, stream>>>(h_na, nodeC, nullptr, Wmu_pr,  bmu_pr,  out + 1 * TS, 1024);
  blocklin_kernel<<<g, 256, 0, stream>>>(h_na, nodeC, nullptr, Wsig_pr, bsig_pr, out + 2 * TS, 1024);
  blocklin_kernel<<<g, 256, 0, stream>>>(h_no, h_na, nodeC,    Wmu_po,  bmu_po,  out + 4 * TS, 1536);
  blocklin_kernel<<<g, 256, 0, stream>>>(h_no, h_na, nodeC,    Wsig_po, bsig_po, out + 5 * TS, 1536);
  // Reparameterization
  int n = (int)TS;
  int blocks = (n + 255) / 256;
  vae_sample_kernel<<<blocks, 256, 0, stream>>>(eps_pr, out + 1 * TS, out + 2 * TS, out + 0 * TS, n);
  vae_sample_kernel<<<blocks, 256, 0, stream>>>(eps_po, out + 4 * TS, out + 5 * TS, out + 3 * TS, n);
}
